// RecurrentTPPResidual_46145128628288
// MI455X (gfx1250) — matrix operand order repaired and checked by reading, not compiled, not run on hardware
//
#include <hip/hip_runtime.h>
#include <math.h>

// ---------------------------------------------------------------------------
// RecurrentTPP residual log-likelihood for MI455X (gfx1250, wave32, WMMA+TDM).
//
// Shapes: B=128, S=4096, D=32 (RNN state), H=192 (MLP width), K=64 (mixture).
// FLOP body = per-token MLP 33->192->192->192  (~84 GFLOP total) -> WMMA f16.
// HBM traffic ~70MB => ~3us @ 23.3 TB/s => compute bound => tensor cores.
// Weights are packed ONCE into WMMA B-fragment order in workspace, then each
// persistent block DMAs the 159KB blob into LDS with TENSOR_LOAD_TO_LDS.
// ---------------------------------------------------------------------------

typedef __attribute__((ext_vector_type(16))) _Float16 v16h;
typedef __attribute__((ext_vector_type(8)))  float    v8f;
typedef __attribute__((ext_vector_type(4)))  unsigned v4u;
typedef __attribute__((ext_vector_type(4)))  int      v4i;
typedef __attribute__((ext_vector_type(8)))  int      v8i;

#define Bn   128
#define Sn   4096
#define Dn   32
#define Hn   192
#define KMIX 64
#define LOG_2PI 1.83787706640934548356f

#define NTILES  ((Bn * Sn) / 64)        // 8192 64-token tiles
#define MLPGRID 1024                    // persistent blocks (divides NTILES)

// packed blob: [w1 6144 f16][w2 36864 f16][w3 36864 f16][w1last|b1|b2|b3 768 f32]
#define W1_F16   6144
#define W23_F16  36864
#define BLOB_F16 (W1_F16 + 2 * W23_F16)             // 79872 f16 = 159744 B
#define BLOB_BYTES (BLOB_F16 * 2 + 768 * 4)         // 162816 B = 20352 x 8B
#define CTX_BYTES ((size_t)Bn * Sn * Dn * 2)        // 32 MB fp16 context

// ---- WMMA 16-bit A-matrix (16xK) lane/element mapping (ISA 7.12.2) --------
// lane L: row m = L&15, sel = L>>4 ; element e -> k = sel*8 + e + (e>=8 ? 8:0)
// Inverse: for k in [0,32): sel = (k>>3)&1, e = (k&7) + (k&16 ? 8 : 0)
__device__ __forceinline__ int a_sel_of_k(int k) { return (k >> 3) & 1; }
__device__ __forceinline__ int a_e_of_k(int k)   { return (k & 7) + ((k & 16) ? 8 : 0); }

// ---------------------------------------------------------------------------
// Kernel 1: gated recurrence.  One wave32 per batch row, lane = state dim d.
// Emits context (= h shifted by one step, context_init at s=0) straight into
// A-fragment layout in workspace: frag tile t/16, addr = tile*512 + lane'*16+e
// ---------------------------------------------------------------------------
__global__ __launch_bounds__(32)
void recur_kernel(const float* __restrict__ it, const float* __restrict__ W,
                  const float* __restrict__ v, const float* __restrict__ b,
                  const float* __restrict__ c0, _Float16* __restrict__ ctxA)
{
    const int d  = threadIdx.x;        // 0..31
    const int bb = blockIdx.x;         // batch row
    const float w0 = W[d], w1 = W[Dn + d], w2 = W[2 * Dn + d], w3 = W[3 * Dn + d];
    const float vf = v[d], vr = v[Dn + d], bf = b[d], br = b[Dn + d];

    const int laneoff = a_sel_of_k(d) * 256 + a_e_of_k(d); // + mrow*16

    float c = 0.0f;
    float hprev = c0[d];               // context_init
    const float* itb = it + (size_t)bb * Sn;
    const size_t tile0 = ((size_t)bb * Sn) >> 4;

    for (int s = 0; s < Sn; ++s) {
        const size_t tile = tile0 + (s >> 4);
        const int mrow = s & 15;
        ctxA[tile * 512 + (size_t)mrow * 16 + laneoff] = (_Float16)hprev;

        const float feat = logf(itb[s] + 1e-8f);
        const float u0 = feat * w0, u1 = feat * w1, u2 = feat * w2, u3 = feat * w3;
        const float f = 1.0f / (1.0f + expf(-(u1 + vf * c + bf)));
        c = f * c + (1.0f - f) * u0;
        const float r = 1.0f / (1.0f + expf(-(u2 + vr * c + br)));
        hprev = r * c + (1.0f - r) * u3;
    }
}

// ---------------------------------------------------------------------------
// Kernel 2: one-time weight pack into WMMA B-fragment order (f16) + f32 tail.
// B 32x16 (KxN), lane L: n = L&15, k = (L>>4)*8 + (e&7) + (e>=8 ? 16:0)
// ---------------------------------------------------------------------------
__global__ __launch_bounds__(256)
void pack_weights_kernel(const float* __restrict__ W1, const float* __restrict__ b1,
                         const float* __restrict__ W2, const float* __restrict__ b2,
                         const float* __restrict__ W3, const float* __restrict__ b3,
                         _Float16* __restrict__ blob16, float* __restrict__ blobf)
{
    const int i = blockIdx.x * blockDim.x + threadIdx.x;
    if (i < W1_F16) {                                   // W1 rows 0..31
        int f = i >> 9, w = i & 511, L = w >> 4, e = w & 15;
        int k = (L >> 4) * 8 + (e & 7) + ((e & 8) ? 16 : 0), n = f * 16 + (L & 15);
        blob16[i] = (_Float16)W1[k * Hn + n];
    } else if (i < W1_F16 + 2 * W23_F16) {              // W2 then W3
        int p = i - W1_F16;
        const float* Wsrc = W2;
        if (p >= W23_F16) { p -= W23_F16; Wsrc = W3; }
        int f = p >> 9, w = p & 511, L = w >> 4, e = w & 15;
        int ch = f % 6, jn = f / 6;
        int k = ch * 32 + (L >> 4) * 8 + (e & 7) + ((e & 8) ? 16 : 0), n = jn * 16 + (L & 15);
        blob16[i] = (_Float16)Wsrc[k * Hn + n];
    } else if (i < BLOB_F16 + 768) {                    // f32 tail
        int j = i - BLOB_F16;
        float val;
        if      (j < 192) val = W1[32 * Hn + j];        // w1last
        else if (j < 384) val = b1[j - 192];
        else if (j < 576) val = b2[j - 384];
        else              val = b3[j - 576];
        blobf[j] = val;
    }
}

__global__ void zero_out_kernel(float* out, int n)
{
    int i = blockIdx.x * blockDim.x + threadIdx.x;
    if (i < n) out[i] = 0.0f;
}

__device__ __forceinline__ float wave_max(float x)
{
    for (int o = 16; o > 0; o >>= 1) x = fmaxf(x, __shfl_xor(x, o, 32));
    return x;
}
__device__ __forceinline__ float wave_sum(float x)
{
    for (int o = 16; o > 0; o >>= 1) x += __shfl_xor(x, o, 32);
    return x;
}

// ---------------------------------------------------------------------------
// TDM: DMA the packed weight blob (1-D tile of n8 8-byte elements) into LDS.
// D# group0: count=1 | lds_addr | global_addr | type=2.  group1: data_size=8B,
// tensor_dim0 = tile_dim0 = n8 (single 1-D tile).  Groups 2/3 zero (<=2-D).
// Tracked by TENSORcnt; EXEC-independent; issue from one wave only.
// ---------------------------------------------------------------------------
__device__ __forceinline__ void tdm_load_blob(const void* gsrc, unsigned lds_byte,
                                              unsigned n8)
{
    unsigned long long ga = (unsigned long long)gsrc;
    v4u g0 = {};
    g0[0] = 1u;                                          // count = 1 (valid)
    g0[1] = lds_byte;                                    // lds_addr (bytes)
    g0[2] = (unsigned)ga;                                // global_addr[31:0]
    g0[3] = (unsigned)((ga >> 32) & 0x01FFFFFFu) | (2u << 30); // [56:32]|type=2
    v8i g1 = {};
    g1[0] = (int)(3u << 16);                             // data_size = 3 (8B)
    g1[1] = (int)((n8 & 0xFFFFu) << 16);                 // tensor_dim0[15:0]
    g1[2] = (int)((n8 >> 16) & 0xFFFFu);                 // tensor_dim0[31:16]
    g1[3] = (int)((n8 & 0xFFFFu) << 16);                 // tile_dim0
    v4i gz = {};
#if __clang_major__ >= 23
    v8i gz8 = {};
    __builtin_amdgcn_tensor_load_to_lds(g0, g1, gz, gz, gz8, 0);
#else
    __builtin_amdgcn_tensor_load_to_lds(g0, g1, gz, gz, 0);
#endif
}

// ---------------------------------------------------------------------------
// Kernel 3: 3-layer MLP with WMMA + mixture log-likelihood reduction.
// Persistent: 1024 blocks x 128 threads (4 waves), each block loops over 8
// 64-token tiles; weights DMA'd once per block into LDS by wave 0 via TDM.
// ---------------------------------------------------------------------------
// LDS layout (bytes), total = 236800 (< 320KB/WGP):
//   [0)       w1s   : 12 frags * 512 f16          = 12288
//   [12288)   w2s   : 72 frags * 512 f16          = 73728
//   [86016)   w3s   : 72 frags * 512 f16          = 73728
//   [159744)  w1last: 192 f32                     = 768
//   [160512)  b1s/b2s/b3s : 3 * 192 f32           = 2304
//   [162816)  agefS : 64 f32 (4 waves * 16 rows)  = 256
//   [163072)  actA  : 4 waves * 6 chunks * 512 f16= 24576
//   [187648)  rawS  : 4 waves * 16 * 192 f32      = 49152
#define MLP_SMEM_BYTES 236800

__global__ __launch_bounds__(128)
void mlp_kernel(const float* __restrict__ it, const float* __restrict__ msk,
                const float* __restrict__ au, const _Float16* __restrict__ ctxA,
                const _Float16* __restrict__ wblob, float* __restrict__ out)
{
    extern __shared__ char smem[];
    _Float16* w1s    = (_Float16*)smem;                   //  6144 f16
    _Float16* w2s    = w1s + 6144;                        // 36864 f16
    _Float16* w3s    = w2s + 36864;                       // 36864 f16
    float*    w1last = (float*)(w3s + 36864);             //   192 f32
    float*    b1s    = w1last + 192;
    float*    b2s    = b1s + 192;
    float*    b3s    = b2s + 192;
    float*    agefS  = b3s + 192;                         //    64 f32
    _Float16* actA   = (_Float16*)(agefS + 64);           // 12288 f16
    float*    rawS   = (float*)(actA + 12288);            // 12288 f32

    const int tid  = threadIdx.x;
    const int lane = tid & 31;
    const int wv   = tid >> 5;
    const int m    = lane & 15;       // C/D column (N), A row (M)
    const int sel  = lane >> 4;

    // ---- DMA full weight blob (f16 frags + f32 tail) into LDS, wave 0 -----
    if (wv == 0) {
        const unsigned ldsbase = (unsigned)__builtin_amdgcn_groupstaticsize();
        tdm_load_blob(wblob, ldsbase, BLOB_BYTES / 8);
        __builtin_amdgcn_s_wait_tensorcnt(0);
    }
    __syncthreads();

    _Float16* myact = actA + wv * 3072;
    float*    myraw = rawS + wv * 16 * Hn;

    for (int tb = blockIdx.x; tb < NTILES; tb += MLPGRID) {
        const size_t tok0 = (size_t)tb * 64 + (size_t)wv * 16;

        if (lane < 16) {                                  // age feature (f32)
            size_t t = tok0 + lane;
            float iv = it[t], av = au[t];
            agefS[wv * 16 + lane] = logf((iv - iv * av) + 1e-8f);
        }
        // per-wave LDS staging: same-wave DS ops complete in order (DScnt),
        // so no block barrier is needed between the phases below.

        // ---- layer 1: h1 = relu(ctx32 @ W1[:32] + agef * W1[32] + b1) -----
        const v16h a1 = *(const v16h*)(ctxA + (tok0 >> 4) * 512 + (size_t)lane * 16);
        for (int jn = 0; jn < 12; ++jn) {
            v8f c;
            const float bias = b1s[jn * 16 + m];
            #pragma unroll
            for (int r = 0; r < 8; ++r) c[r] = bias;
            const v16h bf = *(const v16h*)(w1s + jn * 512 + lane * 16);
            c = __builtin_amdgcn_wmma_f32_16x16x32_f16(false, a1, false, bf,
                                                       (short)0, c, false, false);
            const float wl = w1last[jn * 16 + m];
            const int kcol = jn * 16 + m, ch = kcol >> 5, kk = kcol & 31;
            const int lp = 16 * ((kk >> 3) & 1), e2 = (kk & 7) + ((kk & 16) ? 8 : 0);
            #pragma unroll
            for (int r = 0; r < 8; ++r) {
                const int row = r + 8 * sel;
                const float val = fmaxf(c[r] + agefS[wv * 16 + row] * wl, 0.0f);
                myact[ch * 512 + (row + lp) * 16 + e2] = (_Float16)val;
            }
        }

        // ---- layer 2: h2 = relu(h1 @ W2 + b2) -----------------------------
        v16h a2[6];
        #pragma unroll
        for (int ch = 0; ch < 6; ++ch)
            a2[ch] = *(const v16h*)(myact + ch * 512 + lane * 16);
        for (int jn = 0; jn < 12; ++jn) {
            v8f c;
            const float bias = b2s[jn * 16 + m];
            #pragma unroll
            for (int r = 0; r < 8; ++r) c[r] = bias;
            #pragma unroll
            for (int ch = 0; ch < 6; ++ch) {
                const v16h bf = *(const v16h*)(w2s + (jn * 6 + ch) * 512 + lane * 16);
                c = __builtin_amdgcn_wmma_f32_16x16x32_f16(false, a2[ch], false, bf,
                                                           (short)0, c, false, false);
            }
            const int kcol = jn * 16 + m, ch2 = kcol >> 5, kk = kcol & 31;
            const int lp = 16 * ((kk >> 3) & 1), e2 = (kk & 7) + ((kk & 16) ? 8 : 0);
            #pragma unroll
            for (int r = 0; r < 8; ++r) {
                const int row = r + 8 * sel;
                myact[ch2 * 512 + (row + lp) * 16 + e2] = (_Float16)fmaxf(c[r], 0.0f);
            }
        }

        // ---- layer 3: raw = h2 @ W3 + b3  (no relu) -----------------------
        v16h a3[6];
        #pragma unroll
        for (int ch = 0; ch < 6; ++ch)
            a3[ch] = *(const v16h*)(myact + ch * 512 + lane * 16);
        for (int jn = 0; jn < 12; ++jn) {
            v8f c;
            const float bias = b3s[jn * 16 + m];
            #pragma unroll
            for (int r = 0; r < 8; ++r) c[r] = bias;
            #pragma unroll
            for (int ch = 0; ch < 6; ++ch) {
                const v16h bf = *(const v16h*)(w3s + (jn * 6 + ch) * 512 + lane * 16);
                c = __builtin_amdgcn_wmma_f32_16x16x32_f16(false, a3[ch], false, bf,
                                                           (short)0, c, false, false);
            }
            #pragma unroll
            for (int r = 0; r < 8; ++r)
                myraw[(r + 8 * sel) * Hn + jn * 16 + m] = c[r];
        }

        // ---- mixture log-likelihood, per wave over its 16 rows ------------
        // raw[0:64]=locs, raw[64:128]=log_scales (clamped), raw[128:192]=log_w
        float acc = 0.0f;
        for (int row = 0; row < 16; ++row) {
            const size_t t  = tok0 + row;
            const float* rr = myraw + row * Hn;

            const float l0 = rr[lane],        l1 = rr[32 + lane];
            const float s0 = fminf(fmaxf(rr[64 + lane], -5.0f), 3.0f);
            const float s1 = fminf(fmaxf(rr[96 + lane], -5.0f), 3.0f);
            const float g0 = rr[128 + lane],  g1 = rr[160 + lane];

            const float gm   = wave_max(fmaxf(g0, g1));
            const float lseW = gm + logf(wave_sum(expf(g0 - gm) + expf(g1 - gm)));

            const float iv = it[t], av = au[t];
            const float y  = logf(fmaxf(iv * av, 1e-10f));   // residual_times
            const float z0 = (y - l0) * expf(-s0);
            const float z1 = (y - l1) * expf(-s1);
            const float c0 = (g0 - lseW) - s0 - 0.5f * LOG_2PI - 0.5f * z0 * z0;
            const float c1 = (g1 - lseW) - s1 - 0.5f * LOG_2PI - 0.5f * z1 * z1;

            const float cm = wave_max(fmaxf(c0, c1));
            const float lp = cm + logf(wave_sum(expf(c0 - cm) + expf(c1 - cm))) - y;
            acc += lp * msk[t];
        }
        if (lane == 0) atomicAdd(&out[tok0 >> 12], acc);  // S = 4096 = 2^12
    }
}

// ---------------------------------------------------------------------------
extern "C" void kernel_launch(void* const* d_in, const int* in_sizes, int n_in,
                              void* d_out, int out_size, void* d_ws, size_t ws_size,
                              hipStream_t stream)
{
    (void)in_sizes; (void)n_in; (void)ws_size;
    const float* it  = (const float*)d_in[0];
    const float* msk = (const float*)d_in[1];
    const float* au  = (const float*)d_in[2];
    const float* W   = (const float*)d_in[3];
    const float* v   = (const float*)d_in[4];
    const float* b   = (const float*)d_in[5];
    const float* c0  = (const float*)d_in[6];
    const float* W1  = (const float*)d_in[7];
    const float* b1  = (const float*)d_in[8];
    const float* W2  = (const float*)d_in[9];
    const float* b2  = (const float*)d_in[10];
    const float* W3  = (const float*)d_in[11];
    const float* b3  = (const float*)d_in[12];
    float* out = (float*)d_out;

    // workspace: [ctxA: 32MB f16 A-fragments][packed weight blob: 159KB+3KB]
    _Float16* ctxA   = (_Float16*)d_ws;
    _Float16* blob16 = (_Float16*)((char*)d_ws + CTX_BYTES);
    float*    blobf  = (float*)((char*)d_ws + CTX_BYTES + BLOB_F16 * 2);

    zero_out_kernel<<<1, 128, 0, stream>>>(out, out_size);
    pack_weights_kernel<<<(BLOB_F16 + 768 + 255) / 256, 256, 0, stream>>>(
        W1, b1, W2, b2, W3, b3, blob16, blobf);
    recur_kernel<<<Bn, 32, 0, stream>>>(it, W, v, b, c0, ctxA);
    mlp_kernel<<<MLPGRID, 128, MLP_SMEM_BYTES, stream>>>(
        it, msk, au, ctxA, blob16, out);
}
